// Attention_40226663694919
// MI455X (gfx1250) — matrix operand order repaired and checked by reading, not compiled, not run on hardware
//
#include <hip/hip_runtime.h>

#define DIM   512
#define HEADS 8
#define DH    64
#define BB    4
#define NN    2048
#define BN    (BB * NN)     // 8192 rows
#define NQKV  (3 * DIM)     // 1536
#define ATTN_SCALE 0.125f   // 64^-0.5

typedef __attribute__((ext_vector_type(8)))  __bf16 v8bf;
typedef __attribute__((ext_vector_type(16))) __bf16 v16bf;
typedef __attribute__((ext_vector_type(8)))  float  v8f;

static __device__ __forceinline__ v8f wmma_bf16(v16bf a, v16bf b, v8f c) {
  // D = A(16x32 bf16) * B(32x16 bf16) + C(16x16 f32)
  return __builtin_amdgcn_wmma_f32_16x16x32_bf16(
      /*neg_a=*/false, a, /*neg_b=*/false, b,
      /*c_mod=*/(short)0, c, /*reuse_a=*/false, /*reuse_b=*/false);
}

// Async global->LDS copy, 16 bytes per lane. Tracked by ASYNCcnt (CDNA5 §10.7 /
// opcode 98 GLOBAL_LOAD_ASYNC_TO_LDS_B128). LDS destination address is the low
// 32 bits of the generic shared pointer (flat LDS aperture).
static __device__ __forceinline__ void async_g2l_b128(const __bf16* g, const __bf16* l) {
  unsigned lds = (unsigned)(unsigned long long)l;
  asm volatile("global_load_async_to_lds_b128 %0, %1, off"
               :: "v"(lds), "v"(g) : "memory");
}
static __device__ __forceinline__ void wait_async0() {
  asm volatile("s_wait_asynccnt 0" ::: "memory");
}

static __device__ __forceinline__ v16bf pack16(v8bf lo, v8bf hi) {
  v16bf r;
#pragma unroll
  for (int i = 0; i < 8; ++i) { r[i] = lo[i]; r[8 + i] = hi[i]; }
  return r;
}

// A fragment (16x32 bf16, MxK) from row-major base[m*lda + k].
// Lane L (0..31): M = L&15. Lanes 0-15 hold K = {0..7, 16..23}+k0,
// lanes 16-31 hold K = {8..15, 24..31}+k0  (ISA 16-bit A 16x32 layout).
static __device__ __forceinline__ v16bf load_a(const __bf16* base, int lda,
                                               int m0, int k0, int lane) {
  int m = m0 + (lane & 15);
  int g = (lane >> 4) * 8;
  const __bf16* p = base + (long)m * lda + k0 + g;
  v8bf lo = *(const v8bf*)p;          // K = k0+g .. k0+g+7
  v8bf hi = *(const v8bf*)(p + 16);   // K = k0+16+g .. k0+16+g+7
  return pack16(lo, hi);
}

// B fragment (32x16 bf16, KxN) from K-contiguous layout bt[n*ldb + k].
// Lane L: N = L&15, K = k0 + (L>>4)*8 + {0..7, 16..23} (matches A layout).
// Works for both global and LDS base pointers (compiler picks global/ds loads).
static __device__ __forceinline__ v16bf load_b(const __bf16* bt, int ldb,
                                               int n0, int k0, int lane) {
  int n = n0 + (lane & 15);
  int k = k0 + (lane >> 4) * 8;
  const __bf16* p = bt + (long)n * ldb + k;
  v8bf lo = *(const v8bf*)p;
  v8bf hi = *(const v8bf*)(p + 16);
  return pack16(lo, hi);
}

// ---------------- LayerNorm + bf16 cast: one row (512 f32) per block ----------------
__global__ __launch_bounds__(256) void ln_cast_kernel(
    const float* __restrict__ x, const float* __restrict__ g,
    const float* __restrict__ bia, __bf16* __restrict__ y) {
  int row = blockIdx.x;
  int t = threadIdx.x;
  const float2 v = ((const float2*)(x + (long)row * DIM))[t];
  float s = v.x + v.y;
  float ss = v.x * v.x + v.y * v.y;
#pragma unroll
  for (int off = 16; off >= 1; off >>= 1) {
    s  += __shfl_xor(s,  off, 32);
    ss += __shfl_xor(ss, off, 32);
  }
  __shared__ float sb[8], ssb[8];
  int wid = t >> 5, lane = t & 31;
  if (lane == 0) { sb[wid] = s; ssb[wid] = ss; }
  __syncthreads();
  float S = 0.f, SS = 0.f;
#pragma unroll
  for (int i = 0; i < 8; ++i) { S += sb[i]; SS += ssb[i]; }
  float mu   = S * (1.0f / DIM);
  float var  = SS * (1.0f / DIM) - mu * mu;
  float rstd = rsqrtf(var + 1e-6f);
  float o0 = (v.x - mu) * rstd * g[2 * t]     + bia[2 * t];
  float o1 = (v.y - mu) * rstd * g[2 * t + 1] + bia[2 * t + 1];
  __bf16* yr = y + (long)row * DIM;
  yr[2 * t]     = (__bf16)o0;
  yr[2 * t + 1] = (__bf16)o1;
}

// ---------------- Cast + transpose weights to bf16 (K-contiguous per output col) ----
__global__ __launch_bounds__(256) void cast_w_kernel(
    const float* __restrict__ wqkv, const float* __restrict__ wout,
    __bf16* __restrict__ wqkvT, __bf16* __restrict__ woutT) {
  int i = blockIdx.x * 256 + threadIdx.x;
  if (i < DIM * NQKV) {
    int r = i / NQKV, c = i % NQKV;                 // wqkv[d=r][e=c]
    wqkvT[(long)c * DIM + r] = (__bf16)wqkv[i];
  }
  if (i < DIM * DIM) {
    int r = i / DIM, c = i % DIM;                   // wout[e=r][d=c]
    woutT[(long)c * DIM + r] = (__bf16)wout[i];
  }
}

// ---------------- Shared GEMM body: C[128x64] per block, B panel staged in LDS ------
// btile: 64 output-columns x 512 K, bf16 = 64KB LDS, filled by async b128 copies.
template <typename STORE>
static __device__ __forceinline__ void gemm_body(const __bf16* __restrict__ A,
                                                 const __bf16* __restrict__ BT,
                                                 int n_blk, STORE&& store_cd) {
  __shared__ __align__(16) __bf16 btile[64 * DIM];
  int t = threadIdx.x;
  int lane = t & 31, wid = t >> 5;
  // Stage the whole B panel once: 4096 16B chunks over 256 threads.
#pragma unroll
  for (int i = 0; i < 16; ++i) {
    int c = i * 256 + t;
    int row = c >> 6;            // 64 chunks of 16B per 1KB row
    int off = (c & 63) * 8;      // element offset within row
    async_g2l_b128(BT + (long)(n_blk + row) * DIM + off, &btile[row * DIM + off]);
  }
  wait_async0();
  __syncthreads();

  int m0 = blockIdx.x * 128 + (wid >> 1) * 32;
  int nr = (wid & 1) * 32;       // column offset within the LDS panel
  v8f c[2][2] = {};
  for (int k0 = 0; k0 < DIM; k0 += 32) {
    v16bf a0 = load_a(A, DIM, m0,      k0, lane);
    v16bf a1 = load_a(A, DIM, m0 + 16, k0, lane);
    v16bf b0 = load_b(btile, DIM, nr,      k0, lane);   // ds_load_b128
    v16bf b1 = load_b(btile, DIM, nr + 16, k0, lane);
    c[0][0] = wmma_bf16(a0, b0, c[0][0]);
    c[0][1] = wmma_bf16(a0, b1, c[0][1]);
    c[1][0] = wmma_bf16(a1, b0, c[1][0]);
    c[1][1] = wmma_bf16(a1, b1, c[1][1]);
  }
#pragma unroll
  for (int i = 0; i < 2; ++i)
#pragma unroll
    for (int j = 0; j < 2; ++j)
#pragma unroll
      for (int r = 0; r < 8; ++r) {
        int m = m0 + i * 16 + r + 8 * (lane >> 4);
        int n = n_blk + nr + j * 16 + (lane & 15);
        store_cd(m, n, c[i][j][r]);
      }
}

// ---------------- QKV GEMM: y[8192,512] x wqkv[512,1536] -> per-head Q/K/V^T --------
__global__ __launch_bounds__(256) void qkv_gemm_kernel(
    const __bf16* __restrict__ y, const __bf16* __restrict__ wT,
    __bf16* __restrict__ q, __bf16* __restrict__ k, __bf16* __restrict__ vT) {
  gemm_body(y, wT, blockIdx.y * 64, [&](int m, int n, float val) {
    int bidx = m >> 11, nn = m & (NN - 1);
    int which = n / DIM, inner = n % DIM;
    int h = inner >> 6, dh = inner & 63;
    __bf16 bv = (__bf16)val;
    long hb = (long)bidx * HEADS + h;
    if (which == 0)      q [(hb * NN + nn) * DH + dh] = bv;
    else if (which == 1) k [(hb * NN + nn) * DH + dh] = bv;
    else                 vT[(hb * DH + dh) * NN + nn] = bv;   // transposed V
  });
}

// ---------------- Output projection: ao[8192,512] x w_out[512,512] -> f32 out -------
__global__ __launch_bounds__(256) void out_proj_kernel(
    const __bf16* __restrict__ ao, const __bf16* __restrict__ wT,
    float* __restrict__ out) {
  gemm_body(ao, wT, blockIdx.y * 64, [&](int m, int n, float val) {
    out[(long)m * DIM + n] = val;
  });
}

// ---------------- Flash attention: block = (b,h) x 64 queries, 4 waves x 16q --------
// K/V tiles for the current 64-key block are async-copied into LDS once and
// shared by all 4 waves (4x traffic reduction vs per-wave global fragments).
__global__ __launch_bounds__(128) void attention_kernel(
    const __bf16* __restrict__ q, const __bf16* __restrict__ k,
    const __bf16* __restrict__ vT, __bf16* __restrict__ ao) {
  __shared__ __align__(16) __bf16 ktile[64 * DH];    // [key][dh]  8KB
  __shared__ __align__(16) __bf16 vtile[64 * DH];    // [dh][key]  8KB
  __shared__ __align__(16) __bf16 psm[4][16 * 64];   // per-wave P relayout 8KB
  int t = threadIdx.x;
  int lane = t & 31, wid = t >> 5;
  int bh = blockIdx.x;
  int bidx = bh >> 3, h = bh & 7;
  int q0 = blockIdx.y * 64 + wid * 16;
  const __bf16* qp = q  + (long)bh * NN * DH;
  const __bf16* kp = k  + (long)bh * NN * DH;
  const __bf16* vp = vT + (long)bh * DH * NN;
  v16bf aq0 = load_a(qp, DH, q0, 0,  lane);
  v16bf aq1 = load_a(qp, DH, q0, 32, lane);
  v8f o[4] = {};
  float mrow[8], lrow[8];
#pragma unroll
  for (int r = 0; r < 8; ++r) { mrow[r] = -1e30f; lrow[r] = 0.f; }

  for (int kb = 0; kb < NN; kb += 64) {
    __syncthreads();   // all waves done reading previous K/V tiles
    // K tile: 64 rows x 128B = 512 chunks; V tile: same. 128 threads x 4 each.
#pragma unroll
    for (int i = 0; i < 4; ++i) {
      int c = i * 128 + t;
      int row = c >> 3, off = (c & 7) * 8;
      async_g2l_b128(kp + (long)(kb + row) * DH + off, &ktile[row * DH + off]);
    }
#pragma unroll
    for (int i = 0; i < 4; ++i) {
      int c = i * 128 + t;
      int d = c >> 3, off = (c & 7) * 8;
      async_g2l_b128(vp + (long)d * NN + kb + off, &vtile[d * DH + off]);
    }
    wait_async0();
    __syncthreads();

    // S tile: 16 queries x 64 keys = 4 WMMA C-tiles, K-dim = dh = 64 (2 frags)
    v8f s[4];
#pragma unroll
    for (int kn = 0; kn < 4; ++kn) {
      v8f acc = {};
      acc = wmma_bf16(aq0, load_b(ktile, DH, kn * 16, 0,  lane), acc);
      acc = wmma_bf16(aq1, load_b(ktile, DH, kn * 16, 32, lane), acc);
      s[kn] = acc;
    }
    // Online softmax: rows live in 16-lane halves (C layout); xor 1..8 stays in half
    float corr[8];
#pragma unroll
    for (int r = 0; r < 8; ++r) {
      float mb = -1e30f;
#pragma unroll
      for (int kn = 0; kn < 4; ++kn) mb = fmaxf(mb, s[kn][r] * ATTN_SCALE);
#pragma unroll
      for (int off = 8; off >= 1; off >>= 1) mb = fmaxf(mb, __shfl_xor(mb, off, 32));
      float mnew = fmaxf(mrow[r], mb);
      corr[r] = __expf(mrow[r] - mnew);
      mrow[r] = mnew;
      float rs = 0.f;
#pragma unroll
      for (int kn = 0; kn < 4; ++kn) {
        float p = __expf(s[kn][r] * ATTN_SCALE - mnew);
        s[kn][r] = p;
        rs += p;
      }
#pragma unroll
      for (int off = 8; off >= 1; off >>= 1) rs += __shfl_xor(rs, off, 32);
      lrow[r] = lrow[r] * corr[r] + rs;
    }
    // Relayout P through this wave's private LDS tile (C layout -> A layout)
#pragma unroll
    for (int kn = 0; kn < 4; ++kn)
#pragma unroll
      for (int r = 0; r < 8; ++r) {
        int m   = r + 8 * (lane >> 4);
        int key = kn * 16 + (lane & 15);
        psm[wid][m * 64 + key] = (__bf16)s[kn][r];
      }
#pragma unroll
    for (int dn = 0; dn < 4; ++dn)
#pragma unroll
      for (int r = 0; r < 8; ++r) o[dn][r] *= corr[r];
    const __bf16* pb = &psm[wid][0];
    v16bf ap0 = load_a(pb, 64, 0, 0,  lane);
    v16bf ap1 = load_a(pb, 64, 0, 32, lane);
    // O += P(16x64) * V(64keys x 64dh): K-dim = keys (2 frags), 4 N-tiles of dh
#pragma unroll
    for (int dn = 0; dn < 4; ++dn) {
      o[dn] = wmma_bf16(ap0, load_b(vtile, DH, dn * 16, 0,  lane), o[dn]);
      o[dn] = wmma_bf16(ap1, load_b(vtile, DH, dn * 16, 32, lane), o[dn]);
    }
  }
  // Normalize and store as [b, n, h*64+d] rows for the out-projection GEMM
#pragma unroll
  for (int dn = 0; dn < 4; ++dn)
#pragma unroll
    for (int r = 0; r < 8; ++r) {
      int qrow = q0 + r + 8 * (lane >> 4);
      int d = dn * 16 + (lane & 15);
      float val = o[dn][r] / lrow[r];
      ao[((long)bidx * NN + qrow) * (HEADS * DH) + h * DH + d] = (__bf16)val;
    }
}

extern "C" void kernel_launch(void* const* d_in, const int* in_sizes, int n_in,
                              void* d_out, int out_size, void* d_ws, size_t ws_size,
                              hipStream_t stream) {
  const float* x        = (const float*)d_in[0];
  const float* ln_scale = (const float*)d_in[1];
  const float* ln_bias  = (const float*)d_in[2];
  const float* w_qkv    = (const float*)d_in[3];
  const float* w_out    = (const float*)d_in[4];
  float* out = (float*)d_out;

  char* ws = (char*)d_ws;
  // Workspace layout (bytes):
  __bf16* y     = (__bf16*)(ws + 0);          //  8 MB: LN output bf16 [8192,512]
  __bf16* wqkvT = (__bf16*)(ws + 8388608);    //  1.5 MB: w_qkv^T bf16 [1536,512]
  __bf16* woutT = (__bf16*)(ws + 9961472);    //  0.5 MB: w_out^T bf16 [512,512]
  __bf16* q     = (__bf16*)(ws + 10485760);   //  8 MB: Q [bh, n, dh]
  __bf16* kk    = (__bf16*)(ws + 18874368);   //  8 MB: K [bh, n, dh]
  __bf16* vT    = (__bf16*)(ws + 27262976);   //  8 MB: V^T [bh, dh, n]
  __bf16* ao    = (__bf16*)(ws + 35651584);   //  8 MB: attn out [b, n, inner]

  ln_cast_kernel<<<BN, 256, 0, stream>>>(x, ln_scale, ln_bias, y);
  cast_w_kernel<<<(DIM * NQKV + 255) / 256, 256, 0, stream>>>(w_qkv, w_out, wqkvT, woutT);
  qkv_gemm_kernel<<<dim3(BN / 128, NQKV / 64), 256, 0, stream>>>(y, wqkvT, q, kk, vT);
  attention_kernel<<<dim3(BB * HEADS, NN / 64), 128, 0, stream>>>(q, kk, vT, ao);
  out_proj_kernel<<<dim3(BN / 128, DIM / 64), 256, 0, stream>>>(ao, woutT, out);
}
